// PIGNN_29669634081209
// MI455X (gfx1250) — matrix operand order repaired and checked by reading, not compiled, not run on hardware
//
#include <hip/hip_runtime.h>
#include <hip/hip_bf16.h>

// ---------------------------------------------------------------------------
// PIGNN on MI455X (gfx1250): bf16 WMMA GEMMs, fused gather->MLP->scatter,
// 4-way M-tile register blocking so each B-fragment load feeds 4 WMMAs.
// ---------------------------------------------------------------------------

typedef __attribute__((ext_vector_type(16))) __bf16 v16bf;
typedef __attribute__((ext_vector_type(8)))  __bf16 v8bf;
typedef __attribute__((ext_vector_type(8)))  float  v8f;

#define HDIM 128
#define NLAYERS 6

// ---- fragment helpers -----------------------------------------------------

static __device__ __forceinline__ v8f zero8() {
  v8f z;
#pragma unroll
  for (int i = 0; i < 8; ++i) z[i] = 0.0f;
  return z;
}

// 16x32 bf16 A-fragment, ISA layout. `pre` = row base + per-lane half offset
// ((lane&16)?8:0) + kt*32. Two contiguous 16B loads per lane.
static __device__ __forceinline__ v16bf a_frag_pre(const __bf16* pre) {
  v8bf c0 = *(const v8bf*)(pre);
  v8bf c1 = *(const v8bf*)(pre + 16);
  v16bf r;
#pragma unroll
  for (int i = 0; i < 8; ++i) { r[i] = c0[i]; r[i + 8] = c1[i]; }
  return r;
}

// fp32-sourced A fragment (converted on the fly), same pre-offset convention
static __device__ __forceinline__ v16bf a_frag_pre_f32(const float* pre) {
  v16bf r;
#pragma unroll
  for (int i = 0; i < 8; ++i) {
    r[i]     = (__bf16)pre[i];
    r[i + 8] = (__bf16)pre[16 + i];
  }
  return r;
}

// 32x16 bf16 B-fragment from pre-packed weights (512 contiguous bf16/frag)
static __device__ __forceinline__ v16bf b_frag(const __bf16* P, int kt, int nt,
                                               int nTiles, int lane) {
  return *(const v16bf*)(P + (((size_t)(kt * nTiles + nt)) << 9) + (lane << 4));
}

static __device__ __forceinline__ v8f wmma_bf16(v16bf a, v16bf b, v8f c) {
  return __builtin_amdgcn_wmma_f32_16x16x32_bf16(false, a, false, b,
                                                 (short)0, c, false, false);
}

// wave-local LDS RAW fence (writes by other lanes of the same wave)
static __device__ __forceinline__ void lds_fence() {
  asm volatile("s_wait_dscnt 0" ::: "memory");
}

// ---- weight repack: fp32 [K,C] row-major -> bf16 B-fragment layout --------
__global__ void repack_kernel(const float* __restrict__ W, __bf16* __restrict__ P,
                              int K, int C) {
  int tid = blockIdx.x * blockDim.x + threadIdx.x;
  if (tid >= K * C) return;
  int frag = tid >> 9;
  int rem  = tid & 511;
  int ln   = rem >> 4;
  int j    = rem & 15;
  int nTiles = C >> 4;
  int kt = frag / nTiles;
  int nt = frag - kt * nTiles;
  int n  = nt * 16 + (ln & 15);
  int k  = kt * 32 + ((ln & 16) ? 8 : 0) + (j & 7) + ((j & 8) ? 16 : 0);
  P[tid] = (__bf16)W[k * C + n];
}

// ---- fused encoder (2-tile blocked): relu(xin@W1+b1) @ W2 + b2 ------------
template <int IN_DIM, bool USE_COORDS>
__global__ __launch_bounds__(64) void encoder_kernel(
    const float* __restrict__ feat, const float* __restrict__ coords,
    const float* __restrict__ W1, const float* __restrict__ b1,
    const __bf16* __restrict__ W2p, const float* __restrict__ b2,
    float* __restrict__ out_f32,   // nullable
    __bf16* __restrict__ out_bf, int M) {
  __shared__ __bf16 smem[2 * 2 * 16 * HDIM];          // 16 KB
  const int wave = threadIdx.x >> 5;
  const int lane = threadIdx.x & 31;
  const int grp  = blockIdx.x * 2 + wave;
  const int t0   = grp * 2;
  if (t0 * 16 >= M) return;
  __bf16* sm = smem + wave * 2 * 16 * HDIM;

  // layer 1 (tiny K) on VALU into LDS, rows clamped for the tail tile
  for (int idx = lane; idx < 2 * 16 * HDIM; idx += 32) {
    const int m2 = idx >> 7;                 // 0..31
    const int n  = idx & (HDIM - 1);
    int row = t0 * 16 + m2;
    if (row >= M) row = M - 1;
    float acc = b1[n];
#pragma unroll
    for (int k = 0; k < IN_DIM; ++k) {
      float xv;
      if (USE_COORDS && k < 3) xv = coords[(size_t)row * 3 + k];
      else                     xv = feat[(size_t)row * IN_DIM + k];
      acc += xv * W1[k * HDIM + n];
    }
    sm[idx] = (__bf16)(acc > 0.0f ? acc : 0.0f);
  }
  lds_fence();

  // layer 2: K=128 WMMA, 2 M-tiles share each B fragment
  const int aoff = (lane & 16) ? 8 : 0;
  const int m    = lane & 15;
  v8f acc[2][8];
#pragma unroll
  for (int i = 0; i < 2; ++i)
#pragma unroll
    for (int nt = 0; nt < 8; ++nt) acc[i][nt] = zero8();
#pragma unroll
  for (int kt = 0; kt < 4; ++kt) {
    v16bf a[2];
#pragma unroll
    for (int i = 0; i < 2; ++i)
      a[i] = a_frag_pre(sm + (i * 16 + m) * HDIM + kt * 32 + aoff);
#pragma unroll
    for (int nt = 0; nt < 8; ++nt) {
      v16bf b = b_frag(W2p, kt, nt, 8, lane);
#pragma unroll
      for (int i = 0; i < 2; ++i) acc[i][nt] = wmma_bf16(a[i], b, acc[i][nt]);
    }
  }
  const int rb = (lane & 16) ? 8 : 0;
  const int c0 = lane & 15;
#pragma unroll
  for (int i = 0; i < 2; ++i) {
    if ((t0 + i) * 16 >= M) continue;
#pragma unroll
    for (int nt = 0; nt < 8; ++nt) {
      const int col = nt * 16 + c0;
      const float bias = b2[col];
#pragma unroll
      for (int r = 0; r < 8; ++r) {
        const size_t idx = (size_t)((t0 + i) * 16 + rb + r) * HDIM + col;
        float v = acc[i][nt][r] + bias;
        out_bf[idx] = (__bf16)v;
        if (out_f32) out_f32[idx] = v;
      }
    }
  }
}

// ---- edge message MLP + scatter-add, 4-tile blocked (64 edges / wave) -----
__global__ __launch_bounds__(64) void edge_msg_kernel(
    const __bf16* __restrict__ h_bf, const __bf16* __restrict__ e_bf,
    const int* __restrict__ ei,                     // [2,E]
    const __bf16* __restrict__ W1p, const float* __restrict__ b1,   // K=384
    const __bf16* __restrict__ W2p, const float* __restrict__ b2,   // K=128
    float* __restrict__ agg, int E) {
  __shared__ __bf16 smem[2 * 4 * 16 * HDIM];        // 32 KB (2 waves x 16 KB)
  const int wave = threadIdx.x >> 5;
  const int lane = threadIdx.x & 31;
  const int grp  = blockIdx.x * 2 + wave;
  const int t0   = grp * 4;
  if (t0 * 16 >= E) return;                         // E/16 divisible by 4
  __bf16* sm = smem + wave * 4 * 16 * HDIM;

  const int aoff = (lane & 16) ? 8 : 0;
  const int m    = lane & 15;

  const __bf16 *rowS[4], *rowD[4], *rowE[4];
#pragma unroll
  for (int i = 0; i < 4; ++i) {
    const int base = (t0 + i) * 16;
    const int s = ei[base + m];
    const int d = ei[(size_t)E + base + m];
    rowS[i] = h_bf + (size_t)s * HDIM + aoff;
    rowD[i] = h_bf + (size_t)d * HDIM + aoff;
    rowE[i] = e_bf + (size_t)(base + m) * HDIM + aoff;
  }

  // GEMM1: K = 3H = 384 (12 k-steps); gathered A, each B frag feeds 4 WMMAs
  v8f acc[4][8];
#pragma unroll
  for (int i = 0; i < 4; ++i)
#pragma unroll
    for (int nt = 0; nt < 8; ++nt) acc[i][nt] = zero8();
#pragma unroll
  for (int kt = 0; kt < 12; ++kt) {
    v16bf a[4];
#pragma unroll
    for (int i = 0; i < 4; ++i) {
      const __bf16* src = (kt < 4) ? (rowS[i] + kt * 32)
                        : (kt < 8) ? (rowD[i] + (kt - 4) * 32)
                                   : (rowE[i] + (kt - 8) * 32);
      a[i] = a_frag_pre(src);
    }
#pragma unroll
    for (int nt = 0; nt < 8; ++nt) {
      v16bf b = b_frag(W1p, kt, nt, 8, lane);
#pragma unroll
      for (int i = 0; i < 4; ++i) acc[i][nt] = wmma_bf16(a[i], b, acc[i][nt]);
    }
  }

  // bias + relu, C-fragments -> LDS (layout conversion for GEMM2)
  const int rb = (lane & 16) ? 8 : 0;
  const int c0 = lane & 15;
#pragma unroll
  for (int i = 0; i < 4; ++i)
#pragma unroll
    for (int nt = 0; nt < 8; ++nt) {
      const float bias = b1[nt * 16 + c0];
#pragma unroll
      for (int r = 0; r < 8; ++r) {
        float v = acc[i][nt][r] + bias;
        sm[(i * 16 + rb + r) * HDIM + nt * 16 + c0] =
            (__bf16)(v > 0.0f ? v : 0.0f);
      }
    }
  lds_fence();

  // GEMM2: K = 128
#pragma unroll
  for (int i = 0; i < 4; ++i)
#pragma unroll
    for (int nt = 0; nt < 8; ++nt) acc[i][nt] = zero8();
#pragma unroll
  for (int kt = 0; kt < 4; ++kt) {
    v16bf a[4];
#pragma unroll
    for (int i = 0; i < 4; ++i)
      a[i] = a_frag_pre(sm + (i * 16 + m) * HDIM + kt * 32 + aoff);
#pragma unroll
    for (int nt = 0; nt < 8; ++nt) {
      v16bf b = b_frag(W2p, kt, nt, 8, lane);
#pragma unroll
      for (int i = 0; i < 4; ++i) acc[i][nt] = wmma_bf16(a[i], b, acc[i][nt]);
    }
  }

  // scatter-add straight from accumulators
#pragma unroll
  for (int i = 0; i < 4; ++i) {
    const int base = (t0 + i) * 16;
    int dn[8];
#pragma unroll
    for (int r = 0; r < 8; ++r) dn[r] = ei[(size_t)E + base + rb + r];
#pragma unroll
    for (int nt = 0; nt < 8; ++nt) {
      const int col = nt * 16 + c0;
      const float bias = b2[col];
#pragma unroll
      for (int r = 0; r < 8; ++r)
        atomicAdd(&agg[(size_t)dn[r] * HDIM + col], acc[i][nt][r] + bias);
    }
  }
}

// ---- node update MLP + residual, 4-tile blocked ---------------------------
__global__ __launch_bounds__(64) void node_upd_kernel(
    float* __restrict__ h_f32, __bf16* __restrict__ h_bf,
    const float* __restrict__ agg,
    const __bf16* __restrict__ W1p, const float* __restrict__ b1,   // K=256
    const __bf16* __restrict__ W2p, const float* __restrict__ b2,   // K=128
    int N) {
  __shared__ __bf16 smem[2 * 4 * 16 * HDIM];        // 32 KB
  const int wave = threadIdx.x >> 5;
  const int lane = threadIdx.x & 31;
  const int grp  = blockIdx.x * 2 + wave;
  const int t0   = grp * 4;
  if (t0 * 16 >= N) return;
  __bf16* sm = smem + wave * 4 * 16 * HDIM;

  const int aoff = (lane & 16) ? 8 : 0;
  const int m    = lane & 15;

  bool act[4];
  const __bf16* rowH[4];
  const float*  rowA[4];
#pragma unroll
  for (int i = 0; i < 4; ++i) {
    const int base = (t0 + i) * 16;
    act[i] = base < N;
    const int r = act[i] ? base + m : m;            // clamp tail to tile 0
    rowH[i] = h_bf + (size_t)r * HDIM + aoff;
    rowA[i] = agg  + (size_t)r * HDIM + aoff;
  }

  v8f acc[4][8];
#pragma unroll
  for (int i = 0; i < 4; ++i)
#pragma unroll
    for (int nt = 0; nt < 8; ++nt) acc[i][nt] = zero8();
#pragma unroll
  for (int kt = 0; kt < 8; ++kt) {                  // K = 2H = 256
    v16bf a[4];
#pragma unroll
    for (int i = 0; i < 4; ++i)
      a[i] = (kt < 4) ? a_frag_pre(rowH[i] + kt * 32)
                      : a_frag_pre_f32(rowA[i] + (kt - 4) * 32);
#pragma unroll
    for (int nt = 0; nt < 8; ++nt) {
      v16bf b = b_frag(W1p, kt, nt, 8, lane);
#pragma unroll
      for (int i = 0; i < 4; ++i) acc[i][nt] = wmma_bf16(a[i], b, acc[i][nt]);
    }
  }

  const int rb = (lane & 16) ? 8 : 0;
  const int c0 = lane & 15;
#pragma unroll
  for (int i = 0; i < 4; ++i)
#pragma unroll
    for (int nt = 0; nt < 8; ++nt) {
      const float bias = b1[nt * 16 + c0];
#pragma unroll
      for (int r = 0; r < 8; ++r) {
        float v = acc[i][nt][r] + bias;
        sm[(i * 16 + rb + r) * HDIM + nt * 16 + c0] =
            (__bf16)(v > 0.0f ? v : 0.0f);
      }
    }
  lds_fence();

#pragma unroll
  for (int i = 0; i < 4; ++i)
#pragma unroll
    for (int nt = 0; nt < 8; ++nt) acc[i][nt] = zero8();
#pragma unroll
  for (int kt = 0; kt < 4; ++kt) {
    v16bf a[4];
#pragma unroll
    for (int i = 0; i < 4; ++i)
      a[i] = a_frag_pre(sm + (i * 16 + m) * HDIM + kt * 32 + aoff);
#pragma unroll
    for (int nt = 0; nt < 8; ++nt) {
      v16bf b = b_frag(W2p, kt, nt, 8, lane);
#pragma unroll
      for (int i = 0; i < 4; ++i) acc[i][nt] = wmma_bf16(a[i], b, acc[i][nt]);
    }
  }

#pragma unroll
  for (int i = 0; i < 4; ++i) {
    if (!act[i]) continue;
#pragma unroll
    for (int nt = 0; nt < 8; ++nt) {
      const int col = nt * 16 + c0;
      const float bias = b2[col];
#pragma unroll
      for (int r = 0; r < 8; ++r) {
        const size_t idx = (size_t)((t0 + i) * 16 + rb + r) * HDIM + col;
        float v = acc[i][nt][r] + bias + h_f32[idx];   // residual
        h_f32[idx] = v;
        h_bf[idx]  = (__bf16)v;
      }
    }
  }
}

// ---- generic WMMA linear (K=128), 2-tile blocked: out = act(A@W+b) --------
template <int NT, bool RELU>
__global__ __launch_bounds__(64) void linear_kernel(
    const __bf16* __restrict__ A,   // [M,128]
    const __bf16* __restrict__ Wp, const float* __restrict__ b,
    __bf16* __restrict__ out,       // [M, NT*16]
    int M) {
  const int wave = threadIdx.x >> 5;
  const int lane = threadIdx.x & 31;
  const int grp  = blockIdx.x * 2 + wave;
  const int t0   = grp * 2;
  if (t0 * 16 >= M) return;

  const int aoff = (lane & 16) ? 8 : 0;
  const int m    = lane & 15;
  bool act[2];
  const __bf16* aRow[2];
#pragma unroll
  for (int i = 0; i < 2; ++i) {
    const int base = (t0 + i) * 16;
    act[i] = base < M;
    aRow[i] = A + (size_t)((act[i] ? base : 0) + m) * HDIM + aoff;
  }

  v8f acc[2][NT];
#pragma unroll
  for (int i = 0; i < 2; ++i)
#pragma unroll
    for (int nt = 0; nt < NT; ++nt) acc[i][nt] = zero8();
#pragma unroll
  for (int kt = 0; kt < 4; ++kt) {
    v16bf a[2];
#pragma unroll
    for (int i = 0; i < 2; ++i) a[i] = a_frag_pre(aRow[i] + kt * 32);
#pragma unroll
    for (int nt = 0; nt < NT; ++nt) {
      v16bf bf = b_frag(Wp, kt, nt, NT, lane);
#pragma unroll
      for (int i = 0; i < 2; ++i) acc[i][nt] = wmma_bf16(a[i], bf, acc[i][nt]);
    }
  }
  const int rb = (lane & 16) ? 8 : 0;
  const int c0 = lane & 15;
  const int ncols = NT * 16;
#pragma unroll
  for (int i = 0; i < 2; ++i) {
    if (!act[i]) continue;
#pragma unroll
    for (int nt = 0; nt < NT; ++nt) {
      const int col = nt * 16 + c0;
      const float bias = b[col];
#pragma unroll
      for (int r = 0; r < 8; ++r) {
        float v = acc[i][nt][r] + bias;
        if (RELU) v = v > 0.0f ? v : 0.0f;
        out[(size_t)((t0 + i) * 16 + rb + r) * ncols + col] = (__bf16)v;
      }
    }
  }
}

// ---- decoder tail: [N,64] @ [64,3] + b3, then BC masking ------------------
__global__ void dec3_kernel(const __bf16* __restrict__ d2,
                            const float* __restrict__ W3,
                            const float* __restrict__ b3,
                            const float* __restrict__ bc_disp,
                            const float* __restrict__ bc_rot,
                            float* __restrict__ pred, int N) {
  int i = blockIdx.x * blockDim.x + threadIdx.x;
  if (i >= N) return;
  float a0 = b3[0], a1 = b3[1], a2 = b3[2];
#pragma unroll 8
  for (int k = 0; k < 64; ++k) {
    float v = (float)d2[(size_t)i * 64 + k];
    a0 += v * W3[k * 3 + 0];
    a1 += v * W3[k * 3 + 1];
    a2 += v * W3[k * 3 + 2];
  }
  float dm = 1.0f - bc_disp[i];
  float rm = 1.0f - bc_rot[i];
  pred[(size_t)i * 3 + 0] = a0 * dm;
  pred[(size_t)i * 3 + 1] = a1 * dm;
  pred[(size_t)i * 3 + 2] = a2 * rm;
}

// ---------------------------------------------------------------------------
extern "C" void kernel_launch(void* const* d_in, const int* in_sizes, int n_in,
                              void* d_out, int out_size, void* d_ws, size_t ws_size,
                              hipStream_t stream) {
  (void)n_in; (void)out_size; (void)ws_size;
  const int N = in_sizes[1] / 3;    // coords [N,3]
  const int E = in_sizes[2] / 10;   // edge_attr [E,10]

  const float* x         = (const float*)d_in[0];
  const float* coords    = (const float*)d_in[1];
  const float* edge_attr = (const float*)d_in[2];
  const int*   ei        = (const int*)d_in[3];
  const float* bc_disp   = (const float*)d_in[4];
  const float* bc_rot    = (const float*)d_in[5];
  const float* enc_nW1   = (const float*)d_in[6];
  const float* enc_nb1   = (const float*)d_in[7];
  const float* enc_nW2   = (const float*)d_in[8];
  const float* enc_nb2   = (const float*)d_in[9];
  const float* enc_eW1   = (const float*)d_in[10];
  const float* enc_eb1   = (const float*)d_in[11];
  const float* enc_eW2   = (const float*)d_in[12];
  const float* enc_eb2   = (const float*)d_in[13];
  const float* mp_eW1    = (const float*)d_in[14];
  const float* mp_eb1    = (const float*)d_in[15];
  const float* mp_eW2    = (const float*)d_in[16];
  const float* mp_eb2    = (const float*)d_in[17];
  const float* mp_nW1    = (const float*)d_in[18];
  const float* mp_nb1    = (const float*)d_in[19];
  const float* mp_nW2    = (const float*)d_in[20];
  const float* mp_nb2    = (const float*)d_in[21];
  const float* dec_W1    = (const float*)d_in[22];
  const float* dec_b1    = (const float*)d_in[23];
  const float* dec_W2    = (const float*)d_in[24];
  const float* dec_b2    = (const float*)d_in[25];
  const float* dec_W3    = (const float*)d_in[26];
  const float* dec_b3    = (const float*)d_in[27];

  // ---- workspace carve (256B aligned) ----
  char* p = (char*)d_ws;
  auto carve = [&](size_t bytes) {
    char* r = p;
    p += (bytes + 255) & ~(size_t)255;
    return r;
  };
  float*  h_f32 = (float*) carve((size_t)N * HDIM * 4);
  __bf16* h_bf  = (__bf16*)carve((size_t)N * HDIM * 2);
  float*  agg   = (float*) carve((size_t)N * HDIM * 4);
  __bf16* e_bf  = (__bf16*)carve((size_t)E * HDIM * 2);
  __bf16* d1_bf = e_bf;                          // decoder aliases dead e_bf
  __bf16* d2_bf = e_bf + (size_t)N * HDIM;
  __bf16* P_encn = (__bf16*)carve((size_t)128 * 128 * 2);
  __bf16* P_ence = (__bf16*)carve((size_t)128 * 128 * 2);
  __bf16* P_eW1  = (__bf16*)carve((size_t)NLAYERS * 384 * 128 * 2);
  __bf16* P_eW2  = (__bf16*)carve((size_t)NLAYERS * 128 * 128 * 2);
  __bf16* P_nW1  = (__bf16*)carve((size_t)NLAYERS * 256 * 128 * 2);
  __bf16* P_nW2  = (__bf16*)carve((size_t)NLAYERS * 128 * 128 * 2);
  __bf16* P_d1   = (__bf16*)carve((size_t)128 * 128 * 2);
  __bf16* P_d2   = (__bf16*)carve((size_t)128 * 64 * 2);

  auto pack = [&](const float* W, __bf16* P, int K, int C) {
    int total = K * C;
    repack_kernel<<<(total + 255) / 256, 256, 0, stream>>>(W, P, K, C);
  };
  pack(enc_nW2, P_encn, 128, 128);
  pack(enc_eW2, P_ence, 128, 128);
  for (int l = 0; l < NLAYERS; ++l) {
    pack(mp_eW1 + (size_t)l * 384 * 128, P_eW1 + (size_t)l * 384 * 128, 384, 128);
    pack(mp_eW2 + (size_t)l * 128 * 128, P_eW2 + (size_t)l * 128 * 128, 128, 128);
    pack(mp_nW1 + (size_t)l * 256 * 128, P_nW1 + (size_t)l * 256 * 128, 256, 128);
    pack(mp_nW2 + (size_t)l * 128 * 128, P_nW2 + (size_t)l * 128 * 128, 128, 128);
  }
  pack(dec_W1, P_d1, 128, 128);
  pack(dec_W2, P_d2, 128, 64);

  const int nodeTiles = (N + 15) / 16;
  const int edgeTiles = (E + 15) / 16;
  const dim3 blk(64);                              // 2 waves per block

  // 4-tile groups (edge/node MP kernels), 2 groups per block
  const int edgeGrp4 = (edgeTiles + 3) / 4;
  const int nodeGrp4 = (nodeTiles + 3) / 4;
  const int edgeBlocks4 = (edgeGrp4 + 1) / 2;
  const int nodeBlocks4 = (nodeGrp4 + 1) / 2;
  // 2-tile groups (encoder / linear kernels), 2 groups per block
  const int nodeGrp2 = (nodeTiles + 1) / 2;
  const int edgeGrp2 = (edgeTiles + 1) / 2;
  const int nodeBlocks2 = (nodeGrp2 + 1) / 2;
  const int edgeBlocks2 = (edgeGrp2 + 1) / 2;

  // encoders (node encoder overwrites x[:,0:3] with coords on the fly)
  encoder_kernel<9,  true ><<<nodeBlocks2, blk, 0, stream>>>(
      x, coords, enc_nW1, enc_nb1, P_encn, enc_nb2, h_f32, h_bf, N);
  encoder_kernel<10, false><<<edgeBlocks2, blk, 0, stream>>>(
      edge_attr, nullptr, enc_eW1, enc_eb1, P_ence, enc_eb2, nullptr, e_bf, E);

  // message-passing layers
  for (int l = 0; l < NLAYERS; ++l) {
    hipMemsetAsync(agg, 0, (size_t)N * HDIM * sizeof(float), stream);
    edge_msg_kernel<<<edgeBlocks4, blk, 0, stream>>>(
        h_bf, e_bf, ei,
        P_eW1 + (size_t)l * 384 * 128, mp_eb1 + (size_t)l * HDIM,
        P_eW2 + (size_t)l * 128 * 128, mp_eb2 + (size_t)l * HDIM,
        agg, E);
    node_upd_kernel<<<nodeBlocks4, blk, 0, stream>>>(
        h_f32, h_bf, agg,
        P_nW1 + (size_t)l * 256 * 128, mp_nb1 + (size_t)l * HDIM,
        P_nW2 + (size_t)l * 128 * 128, mp_nb2 + (size_t)l * HDIM,
        N);
  }

  // decoder
  linear_kernel<8, true><<<nodeBlocks2, blk, 0, stream>>>(h_bf,  P_d1, dec_b1, d1_bf, N);
  linear_kernel<4, true><<<nodeBlocks2, blk, 0, stream>>>(d1_bf, P_d2, dec_b2, d2_bf, N);
  dec3_kernel<<<(N + 255) / 256, 256, 0, stream>>>(
      d2_bf, dec_W3, dec_b3, bc_disp, bc_rot, (float*)d_out, N);
}
